// LocallyConnected2d_58171037057775
// MI455X (gfx1250) — matrix operand order repaired and checked by reading, not compiled, not run on hardware
//
#include <hip/hip_runtime.h>

// LocallyConnected2d: out[b,o,h,w] = sum_{c,k} x_pad[b,c,h+dh(k),w+dw(k)] * W[o,c,h,w,k]
// B=128, C_IN=C_OUT=64, H=W=32, 3x3 taps. One workgroup per (h,w): GEMM 128x64x576,
// decomposed as 3 tap-rows x 2 channel-halves x 3 taps of K=32 WMMA steps.
// f32 -> f16 convert on the fly (v_cvt_pk_rtz_f16_f32), f32 WMMA accumulate.

typedef __fp16   pkh2 __attribute__((ext_vector_type(2)));   // return type of cvt_pkrtz
typedef _Float16 v8h  __attribute__((ext_vector_type(8)));
typedef _Float16 v16h __attribute__((ext_vector_type(16)));
typedef float    v8f  __attribute__((ext_vector_type(8)));
typedef float    f4   __attribute__((ext_vector_type(4)));

#define LDT 40   // halfs per LDS row: 32-channel tile + 8 pad (16B aligned rows, conflict-free)

__device__ __forceinline__ float selq(f4 v, int idx) {
    // idx is block-uniform (scalar); lowers to cndmask chain
    float r = v.x;
    r = (idx == 1) ? v.y : r;
    r = (idx == 2) ? v.z : r;
    r = (idx == 3) ? v.w : r;
    return r;
}

__global__ __launch_bounds__(256)
void lc2d_wmma_f16(const float* __restrict__ x,
                   const float* __restrict__ wt,
                   float* __restrict__ out)
{
    __shared__ __align__(16) _Float16 As3[3][128][LDT]; // per-tap A plane: [q][b][c_local]
    __shared__ __align__(16) _Float16 Bs3[3][64][LDT];  // per-tap B plane: [q][o][c_local]

    const int hw   = blockIdx.x;       // spatial position 0..1023
    const int h    = hw >> 5;
    const int w    = hw & 31;
    const int tid  = threadIdx.x;
    const int wave = tid >> 5;         // 8 wave32 waves
    const int lane = tid & 31;
    const int nl   = lane & 15;
    const int hi   = lane >> 4;

    // block-uniform horizontal load window: one float4 covers taps q=0..2
    const int wbase = (w == 0) ? 0 : ((w == 31) ? 28 : (w - 1));

    v8f acc[4];
    #pragma unroll
    for (int j = 0; j < 4; ++j) acc[j] = (v8f){0.f,0.f,0.f,0.f,0.f,0.f,0.f,0.f};

    for (int p = 0; p < 3; ++p) {                 // tap row
        const int  y  = h + p - 1;
        const bool yv = ((unsigned)y < 32u);
        const int  yc = yv ? y : (y < 0 ? 0 : 31);    // clamped for addressing only
        const int  abase = (yc << 5) + wbase;
        int  offq[3];
        bool vq[3];
        #pragma unroll
        for (int q = 0; q < 3; ++q) {
            const int col = w + q - 1;
            vq[q] = yv && ((unsigned)col < 32u);
            int o_ = col - wbase;
            offq[q] = (o_ < 0) ? 0 : ((o_ > 3) ? 3 : o_);
        }
        const int kb    = (p == 2) ? 5 : 3 * p;   // shifted window for p=2 (stay in bounds)
        const int wboff = hw * 9 + kb;

        for (int ch = 0; ch < 64; ch += 32) {     // channel half
            // ---- Stage A: 128 rows x 16 channel-pairs; one f4 load spans 3 taps ----
            #pragma unroll
            for (int i = 0; i < 8; ++i) {
                const int e  = tid + (i << 8);    // 0..2047
                const int b  = e >> 4;
                const int cp = e & 15;
                const int c  = ch + (cp << 1);
                const float* px = x + (((b << 6) + c) << 10) + abase;
                f4 v0, v1;
                __builtin_memcpy(&v0, px,        16);   // unaligned b128
                __builtin_memcpy(&v1, px + 1024, 16);   // next channel
                #pragma unroll
                for (int q = 0; q < 3; ++q) {
                    const float a0 = vq[q] ? selq(v0, offq[q]) : 0.0f;
                    const float a1 = vq[q] ? selq(v1, offq[q]) : 0.0f;
                    *(pkh2*)&As3[q][b][cp << 1] = __builtin_amdgcn_cvt_pkrtz(a0, a1);
                }
            }
            // ---- Stage B: 64 rows x 16 channel-pairs; one f4 load spans 3 taps ----
            #pragma unroll
            for (int i = 0; i < 4; ++i) {
                const int e  = tid + (i << 8);    // 0..1023
                const int o  = e >> 4;
                const int cp = e & 15;
                const int c  = ch + (cp << 1);
                const float* pw = wt + ((o << 6) + c) * 9216 + wboff;
                f4 w0, w1;
                __builtin_memcpy(&w0, pw,        16);
                __builtin_memcpy(&w1, pw + 9216, 16);
                #pragma unroll
                for (int q = 0; q < 3; ++q) {
                    const int idx = 3 * p + q - kb;     // 0..2 (p<2) or 1..3 (p==2)
                    *(pkh2*)&Bs3[q][o][cp << 1] =
                        __builtin_amdgcn_cvt_pkrtz(selq(w0, idx), selq(w1, idx));
                }
            }
            __syncthreads();

            // ---- Compute: 3 taps x 4 N-tiles of v_wmma_f32_16x16x32_f16 ----
            #pragma unroll
            for (int q = 0; q < 3; ++q) {
                // A fragment: lane M = nl; elem j -> K = (j&7) + 8*hi + 16*(j>>3)
                v16h a;
                {
                    const v8h a0 = *(const v8h*)&As3[q][(wave << 4) + nl][(hi << 3)];
                    const v8h a1 = *(const v8h*)&As3[q][(wave << 4) + nl][(hi << 3) + 16];
                    #pragma unroll
                    for (int t = 0; t < 8; ++t) { a[t] = a0[t]; a[8 + t] = a1[t]; }
                }
                #pragma unroll
                for (int j = 0; j < 4; ++j) {
                    // B fragment: lane N = nl; elem j -> K = (j&7) + 8*hi + 16*(j>>3)
                    const v8h b0 = *(const v8h*)&Bs3[q][(j << 4) + nl][(hi << 3)];
                    const v8h b1 = *(const v8h*)&Bs3[q][(j << 4) + nl][(hi << 3) + 16];
                    v16h bf;
                    #pragma unroll
                    for (int t = 0; t < 8; ++t) { bf[t] = b0[t]; bf[8 + t] = b1[t]; }
                    acc[j] = __builtin_amdgcn_wmma_f32_16x16x32_f16(
                        false, a, false, bf, (short)0, acc[j], false, false);
                }
            }
            __syncthreads();
        }
    }

    // ---- Epilogue: C/D layout -> VGPR r holds M = r + 8*hi, N = nl ----
    #pragma unroll
    for (int j = 0; j < 4; ++j) {
        #pragma unroll
        for (int r = 0; r < 8; ++r) {
            const int b = (wave << 4) + (hi << 3) + r;
            const int o = (j << 4) + nl;
            out[(((b << 6) + o) << 10) + hw] = acc[j][r];
        }
    }
}

extern "C" void kernel_launch(void* const* d_in, const int* in_sizes, int n_in,
                              void* d_out, int out_size, void* d_ws, size_t ws_size,
                              hipStream_t stream) {
    (void)in_sizes; (void)n_in; (void)out_size; (void)d_ws; (void)ws_size;
    const float* x  = (const float*)d_in[0];   // [128,64,32,32] f32
    const float* wt = (const float*)d_in[1];   // [1,64,64,32,32,9] f32
    float* out = (float*)d_out;                // [128,64,32,32] f32
    lc2d_wmma_f16<<<dim3(32 * 32), dim3(256), 0, stream>>>(x, wt, out);
}
